// EvoformerBlock_23476291239971
// MI455X (gfx1250) — matrix operand order repaired and checked by reading, not compiled, not run on hardware
//
#include <hip/hip_runtime.h>
#include <hip/hip_bf16.h>
#include <math.h>

// ============================================================================
// Evoformer block (AlphaFold-style) for AMD MI455X / gfx1250 (wave32 + WMMA).
//
// Roofline: ~120 GFLOP of dense contraction vs ~70MB unique traffic -> heavily
// compute bound; all contractions run on v_wmma_f32_16x16x32_f16 (every K dim
// here is a multiple of 32; f16 operands / f32 accumulate). Softmax/LN/sigmoid
// stay in f32 VALU.
//
// Data movement strategy:
//  * Generic GEMM: block-cooperative 128x64 macro-tile (8 waves x 16 rows,
//    shared 64-col B panel). B chunks (64x32 f16 = 4KB) are staged in LDS with
//    GLOBAL_LOAD_ASYNC_TO_LDS_B128 (ASYNCcnt), double-buffered so the async
//    copy of chunk k+32 overlaps the 4 WMMAs on chunk k. B fragments then are
//    aligned 32B LDS reads shared by all 8 waves (8x L2-traffic reduction).
//  * 1x4 register blocking everywhere: one A fragment feeds 4 WMMAs
//    (2 for AV, 8 for OPM phase 1), amortizing the strided operand fetch.
//  * A fragments load as aligned float4 pairs (two 32B runs per lane).
//  * Weights repacked once per launch to f16 [n][K] (K-contiguous per output
//    column, zero-padded to 64 columns for the cooperative copy).
//  * __builtin_prefetch (-> global_prefetch_b8) two K-blocks ahead on A.
//  * WMMA reuse_a/b hints intentionally 0: only legal between identical
//    adjacent instructions, which cannot be guaranteed from HIP.
//
// Input pointer layout assumption (compile-only, unverifiable):
//   d_in[0..3] = msa_act, pair_act, msa_mask, pair_mask (insertion order)
//   d_in[4..96] = 93 param leaves in JAX pytree order (keys sorted
//   alphabetically at every level: col, msa_trans, opm, pair_trans, row,
//   tae, tas, tmi, tmo; leaves sorted within each).
//
// d_out = [msa (S*R*CM) | pair (R*R*CZ)] f32, used directly as residual
// accumulators. Workspace requirement ~275 MB.
// ============================================================================

typedef __attribute__((ext_vector_type(16))) _Float16 v16h;
typedef __attribute__((ext_vector_type(8)))  float    v8f;

static constexpr int S_ = 128, R_ = 256, CM_ = 256, CZ_ = 128;
static constexpr int HM_ = 8, HZ_ = 4, D_ = 32, CT_ = 128;
static constexpr size_t MS_ = (size_t)S_ * R_ * CM_;   // 8388608
static constexpr size_t PS_ = (size_t)R_ * R_ * CZ_;   // 8388608
static constexpr size_t SR_ = (size_t)S_ * R_;         // 32768
static constexpr size_t RR_ = (size_t)R_ * R_;         // 65536

// workspace float offsets
static constexpr size_t OFF_X  = 0;
static constexpr size_t OFF_Z  = MS_;
static constexpr size_t OFF_Q  = 2 * MS_;
static constexpr size_t OFF_K  = 3 * MS_;
static constexpr size_t OFF_V  = 4 * MS_;
static constexpr size_t OFF_G  = 5 * MS_;
static constexpr size_t OFF_O  = 6 * MS_;
static constexpr size_t OFF_L  = 7 * MS_;
static constexpr size_t OFF_H  = OFF_Q;            // transition hidden aliases Q..G
static constexpr size_t OFF_NB = 8 * MS_;          // 524288 floats
static constexpr size_t OFF_MT = OFF_NB + 524288;  // 65536 floats
static constexpr size_t OFF_PK = OFF_MT + 65536;   // f16 packed weights (~2M halves)

// ---------------------------------------------------------------------------
// WMMA fragment maps used here (16-bit A 16x32 / B 32x16, f32 C/D; A/B K order chosen by this file)
// A: lane row m = lane&15; element e -> K = (lane>>4)*16 + e
// B: lane col n = lane&15; element e -> K = (lane>>4)*16 + e
// C/D: lane col n = lane&15; vgpr r  -> row m = r + 8*(lane>>4)
// ---------------------------------------------------------------------------
__device__ __forceinline__ int kmapA(int e, int half) { return (half << 4) + e; }
__device__ __forceinline__ int kmapB(int e, int half) { return (half << 4) + e; }

__device__ __forceinline__ v8f wmma16(v16h a, v16h b, v8f c) {
#if defined(__HIP_DEVICE_COMPILE__)
  return __builtin_amdgcn_wmma_f32_16x16x32_f16(false, a, false, b, (short)0, c, false, false);
#else
  return c;
#endif
}

// --- CDNA5 async global->LDS copy (VGLOBAL op 98, tracked by ASYNCcnt) -----
__device__ __forceinline__ void async_b128(unsigned lds_byte_off, const void* gaddr) {
#if defined(__HIP_DEVICE_COMPILE__)
  asm volatile("global_load_async_to_lds_b128 %0, %1, off"
               :: "v"(lds_byte_off), "v"(gaddr) : "memory");
#endif
}
__device__ __forceinline__ void async_wait0() {
#if defined(__HIP_DEVICE_COMPILE__)
  asm volatile("s_wait_asynccnt 0x0" ::: "memory");
#endif
}

// A fragment from a 32-float-aligned f32 row: 16 consecutive K values per lane.
__device__ __forceinline__ v16h loadA16_f32(const float* __restrict__ row, int k0, int half) {
  const float4* p0 = reinterpret_cast<const float4*>(row + k0 + (half << 4));
  const float4* p1 = reinterpret_cast<const float4*>(row + k0 + (half << 4) + 8);
  float4 x0 = p0[0], x1 = p0[1], y0 = p1[0], y1 = p1[1];
  v16h a;
  a[0] = (_Float16)x0.x; a[1] = (_Float16)x0.y; a[2]  = (_Float16)x0.z; a[3]  = (_Float16)x0.w;
  a[4] = (_Float16)x1.x; a[5] = (_Float16)x1.y; a[6]  = (_Float16)x1.z; a[7]  = (_Float16)x1.w;
  a[8] = (_Float16)y0.x; a[9] = (_Float16)y0.y; a[10] = (_Float16)y0.z; a[11] = (_Float16)y0.w;
  a[12] = (_Float16)y1.x; a[13] = (_Float16)y1.y; a[14] = (_Float16)y1.z; a[15] = (_Float16)y1.w;
  return a;
}
// B fragment from a contiguous f32 row (16 consecutive K values per lane).
__device__ __forceinline__ v16h loadB16_f32(const float* __restrict__ row, int half) {
  const float4* p = reinterpret_cast<const float4*>(row + (half << 4));
  float4 x0 = p[0], x1 = p[1], x2 = p[2], x3 = p[3];
  v16h b;
  b[0] = (_Float16)x0.x; b[1] = (_Float16)x0.y; b[2]  = (_Float16)x0.z; b[3]  = (_Float16)x0.w;
  b[4] = (_Float16)x1.x; b[5] = (_Float16)x1.y; b[6]  = (_Float16)x1.z; b[7]  = (_Float16)x1.w;
  b[8] = (_Float16)x2.x; b[9] = (_Float16)x2.y; b[10] = (_Float16)x2.z; b[11] = (_Float16)x2.w;
  b[12] = (_Float16)x3.x; b[13] = (_Float16)x3.y; b[14] = (_Float16)x3.z; b[15] = (_Float16)x3.w;
  return b;
}

__device__ __forceinline__ float wave_sum(float v) {
#pragma unroll
  for (int o = 16; o > 0; o >>= 1) v += __shfl_xor(v, o, 32);
  return v;
}
__device__ __forceinline__ float wave_max(float v) {
#pragma unroll
  for (int o = 16; o > 0; o >>= 1) v = fmaxf(v, __shfl_xor(v, o, 32));
  return v;
}

// ------------------------------- LayerNorm ---------------------------------
__global__ void k_layernorm(const float* __restrict__ x, const float* __restrict__ g,
                            const float* __restrict__ b, float* __restrict__ y,
                            int rows, int C) {
  int row = blockIdx.x * blockDim.y + threadIdx.y;
  if (row >= rows) return;
  const float* xr = x + (size_t)row * C;
  float s = 0.f, s2 = 0.f;
  for (int c = threadIdx.x; c < C; c += 32) { float v = xr[c]; s += v; s2 += v * v; }
  s = wave_sum(s); s2 = wave_sum(s2);
  float mu = s / C;
  float var = s2 / C - mu * mu;
  float rs = rsqrtf(var + 1e-5f);
  float* yr = y + (size_t)row * C;
  for (int c = threadIdx.x; c < C; c += 32) yr[c] = (xr[c] - mu) * rs * g[c] + b[c];
}

// ------------------------- weight pack f32 -> f16 --------------------------
// Bp layout: [n_padded][Kd] halves (K-contiguous per output column), so a
// WMMA B-fragment is one aligned 32B v16h load per lane. Zero-padded to 64
// columns so the GEMM's cooperative 64-row async copy stays in bounds.
__global__ void k_pack(const float* __restrict__ B, _Float16* __restrict__ Bp,
                       int Kd, int N, int Npad) {
  int idx = blockIdx.x * blockDim.x + threadIdx.x;
  if (idx >= Npad * Kd) return;
  int k = idx % Kd;
  int n = idx / Kd;
  Bp[idx] = (n < N) ? (_Float16)B[(size_t)k * N + n] : (_Float16)0.f;
}

// ----------------------------- generic GEMM --------------------------------
// C = epi( act( alpha*(A@B) + bias ) * rowscale )
// A f32 row-major [M,K] (M multiple of 128); Bp packed f16 [n64][K].
// Block = 8 waves: 128x64 macro-tile. The shared 64x32 B chunk is staged in
// LDS by one global_load_async_to_lds_b128 per thread, double-buffered.
// ACT: 0 none, 1 relu, 2 sigmoid.  EPI: 0 store, 1 +=, 2 *=
template <int ACT, int EPI>
__global__ void k_gemm(const float* __restrict__ A, const _Float16* __restrict__ Bp,
                       const float* __restrict__ bias, const float* __restrict__ rowscale,
                       float* __restrict__ C, float alpha, int M, int N, int Kd) {
  __shared__ __align__(64) _Float16 Blds[2][64 * 32];  // 2 x 4KB
  int Ntiles = (N + 15) >> 4;
  int Ngrp = (Ntiles + 3) >> 2;
  int grp = blockIdx.x;
  if (grp >= (M >> 7) * Ngrp) return;  // block-uniform
  int mt = (grp / Ngrp) * 8 + threadIdx.y;
  int nt0 = (grp % Ngrp) * 4;
  int nsub = Ntiles - nt0; if (nsub > 4) nsub = 4;
  int lane = threadIdx.x, half = lane >> 4, l16 = lane & 15;
  int tid = threadIdx.y * 32 + lane;
  const float* Arow = A + (size_t)((mt << 4) + l16) * Kd;
  // cooperative B copy: thread tid moves 16B: row (n) = tid>>2, piece = tid&3
  const _Float16* gsrc = Bp + (size_t)((nt0 << 4) + (tid >> 2)) * Kd + ((tid & 3) << 3);
  unsigned lds0 = (unsigned)(uintptr_t)(&Blds[0][0]) + (unsigned)tid * 16u;
  unsigned lds1 = (unsigned)(uintptr_t)(&Blds[1][0]) + (unsigned)tid * 16u;
  async_b128(lds0, gsrc);
  async_wait0();
  __syncthreads();
  v8f acc[4] = {};
  int cur = 0;
  for (int k0 = 0; k0 < Kd; k0 += 32) {
    if (k0 + 32 < Kd) async_b128(cur ? lds0 : lds1, gsrc + k0 + 32);
    v16h a = loadA16_f32(Arow, k0, half);
    if (k0 + 64 < Kd) __builtin_prefetch((const void*)(Arow + k0 + 64), 0, 3);
#pragma unroll
    for (int s = 0; s < 4; s++) {
      if (s < nsub) {  // wave-uniform branch: EXEC stays all-ones for WMMA
        const _Float16* bp = &Blds[cur][(((s << 4) + l16) << 5) + (half << 4)];
        v16h b = *reinterpret_cast<const v16h*>(bp);
        acc[s] = wmma16(a, b, acc[s]);
      }
    }
    async_wait0();     // next chunk landed
    __syncthreads();   // and everyone is done reading the current one
    cur ^= 1;
  }
#pragma unroll
  for (int s = 0; s < 4; s++) {
    int n = ((nt0 + s) << 4) + l16;
    if (s < nsub && n < N) {
      float bi = bias ? bias[n] : 0.f;
#pragma unroll
      for (int r = 0; r < 8; r++) {
        int mr = (mt << 4) + r + 8 * half;
        float v = acc[s][r] * alpha + bi;
        if (ACT == 1) v = fmaxf(v, 0.f);
        if (ACT == 2) v = 1.f / (1.f + __expf(-v));
        if (rowscale) v *= rowscale[mr];
        size_t o = (size_t)mr * N + n;
        if (EPI == 0) C[o] = v;
        else if (EPI == 1) C[o] += v;
        else C[o] *= v;
      }
    }
  }
}

// ------------------------- attention: Q @ K^T ------------------------------
// Q,K: f32 [Bn*N, H*32]; L chunk: [Bc,H,N,N]. One Q fragment -> 4 key tiles.
__global__ void k_qkt(const float* __restrict__ Q, const float* __restrict__ Km,
                      float* __restrict__ L, int bStart, int Bc, int N, int H, int HD) {
  int nt16 = N >> 4, kg = nt16 >> 2;
  int tile = blockIdx.x * blockDim.y + threadIdx.y;
  if (tile >= Bc * H * nt16 * kg) return;
  int kt0 = (tile % kg) * 4; int t1 = tile / kg;
  int qt = t1 % nt16;        int t2 = t1 / nt16;
  int h = t2 % H;            int b = t2 / H;
  int lane = threadIdx.x, half = lane >> 4, l16 = lane & 15;
  const float* Qr = Q + (size_t)((bStart + b) * N + (qt << 4) + l16) * HD + h * 32;
  v16h a = loadA16_f32(Qr, 0, half);
  v8f acc[4] = {};
#pragma unroll
  for (int s = 0; s < 4; s++) {
    const float* Kr = Km + (size_t)((bStart + b) * N + ((kt0 + s) << 4) + l16) * HD + h * 32;
    v16h bf = loadB16_f32(Kr, half);
    acc[s] = wmma16(a, bf, acc[s]);
  }
  float* Lb = L + ((size_t)(b * H + h) * N + (qt << 4)) * N;
#pragma unroll
  for (int s = 0; s < 4; s++)
#pragma unroll
    for (int r = 0; r < 8; r++)
      Lb[(size_t)(r + 8 * half) * N + ((kt0 + s) << 4) + l16] = acc[s][r];
}

// ------------------------- softmax with biases -----------------------------
// row = (b,h,q); adds 1e9*(mask[b,k]-1) and optional nb[(q*N+k)*H + h].
__global__ void k_softmax(float* __restrict__ L, const float* __restrict__ mask,
                          const float* __restrict__ nb, int bStart, int Bc, int N, int H) {
  int row = blockIdx.x * blockDim.y + threadIdx.y;
  if (row >= Bc * H * N) return;
  int q = row % N; int t = row / N; int h = t % H; int b = t / H;
  float* Lr = L + (size_t)row * N;
  const float* mrow = mask + (size_t)(bStart + b) * N;
  const float* nbr = nb ? nb + (size_t)q * N * H + h : nullptr;
  float mx = -1e30f;
  for (int k = threadIdx.x; k < N; k += 32) {
    float v = Lr[k] + 1e9f * (mrow[k] - 1.f);
    if (nbr) v += nbr[(size_t)k * H];
    Lr[k] = v;
    mx = fmaxf(mx, v);
  }
  mx = wave_max(mx);
  float s = 0.f;
  for (int k = threadIdx.x; k < N; k += 32) { float e = __expf(Lr[k] - mx); Lr[k] = e; s += e; }
  s = wave_sum(s);
  float inv = 1.f / s;
  for (int k = threadIdx.x; k < N; k += 32) Lr[k] *= inv;
}

// ------------------------- attention: W @ V --------------------------------
// One W fragment feeds both d-tiles (D=32).
__global__ void k_av(const float* __restrict__ L, const float* __restrict__ V,
                     float* __restrict__ O, int bStart, int Bc, int N, int H, int HD) {
  int nt16 = N >> 4;
  int tile = blockIdx.x * blockDim.y + threadIdx.y;
  if (tile >= Bc * H * nt16) return;
  int qt = tile % nt16; int t2 = tile / nt16;
  int h = t2 % H;       int b = t2 / H;
  int lane = threadIdx.x, half = lane >> 4, l16 = lane & 15;
  const float* Wr = L + ((size_t)(b * H + h) * N + (qt << 4) + l16) * N;
  const float* Vb0 = V + (size_t)(bStart + b) * N * HD + h * 32 + l16;
  v8f acc[2] = {};
  for (int k0 = 0; k0 < N; k0 += 32) {
    v16h a = loadA16_f32(Wr, k0, half);
#pragma unroll
    for (int dt = 0; dt < 2; dt++) {
      v16h bf;
#pragma unroll
      for (int e = 0; e < 16; e++)
        bf[e] = (_Float16)Vb0[(size_t)(k0 + (half << 4) + e) * HD + dt * 16];
      acc[dt] = wmma16(a, bf, acc[dt]);
    }
  }
  float* Ob = O + (size_t)((bStart + b) * N + (qt << 4)) * HD + h * 32 + l16;
#pragma unroll
  for (int dt = 0; dt < 2; dt++)
#pragma unroll
    for (int r = 0; r < 8; r++)
      Ob[(size_t)(r + 8 * half) * HD + dt * 16] = acc[dt][r];
}

// ------------------------- transposes & elementwise ------------------------
__global__ void k_transpose01(const float* __restrict__ src, float* __restrict__ dst,
                              int A, int B, int C) {  // dst[b,a,c] = src[a,b,c]
  size_t idx = (size_t)blockIdx.x * blockDim.x + threadIdx.x;
  if (idx >= (size_t)A * B * C) return;
  int c = idx % C; size_t t = idx / C; int b = t % B; int a = (int)(t / B);
  dst[((size_t)b * A + a) * C + c] = src[idx];
}
__global__ void k_transpose_add(const float* __restrict__ src, float* __restrict__ dst,
                                int A, int B, int C) {  // dst[a,b,c] += src[b,a,c]
  size_t idx = (size_t)blockIdx.x * blockDim.x + threadIdx.x;
  if (idx >= (size_t)A * B * C) return;
  int c = idx % C; size_t t = idx / C; int b = t % B; int a = (int)(t / B);
  dst[idx] += src[((size_t)b * A + a) * C + c];
}
__global__ void k_mul(float* __restrict__ d, const float* __restrict__ s, size_t n) {
  size_t i = (size_t)blockIdx.x * blockDim.x + threadIdx.x;
  if (i < n) d[i] *= s[i];
}
__global__ void k_fma_add(float* __restrict__ d, const float* __restrict__ a,
                          const float* __restrict__ b, size_t n) {
  size_t i = (size_t)blockIdx.x * blockDim.x + threadIdx.x;
  if (i < n) d[i] += a[i] * b[i];
}

// ---------------- triangle-mult channel-wise batched GEMM ------------------
// outgoing: o[i,j,c] = sum_k L[i,k,c]*R[j,k,c];  incoming: sum_k L[k,i,c]*R[k,j,c]
// One A fragment -> 4 j-tiles.
__global__ void k_trimul_o(const float* __restrict__ Lt, const float* __restrict__ Rt,
                           float* __restrict__ Ot, int Rn, int Ck, int outgoing) {
  int c = blockIdx.z, it = blockIdx.x, jt0 = blockIdx.y * 4;
  int lane = threadIdx.x, half = lane >> 4, l16 = lane & 15;
  v8f acc[4] = {};
  for (int k0 = 0; k0 < Rn; k0 += 32) {
    v16h a;
#pragma unroll
    for (int e = 0; e < 16; e++) {
      int ka = k0 + kmapA(e, half);
      size_t ia = outgoing ? ((size_t)((it * 16 + l16) * Rn + ka) * Ck + c)
                           : ((size_t)(ka * Rn + it * 16 + l16) * Ck + c);
      a[e] = (_Float16)Lt[ia];
    }
#pragma unroll
    for (int s = 0; s < 4; s++) {
      int j = (jt0 + s) * 16 + l16;
      v16h b;
#pragma unroll
      for (int e = 0; e < 16; e++) {
        int kb = k0 + kmapB(e, half);
        size_t ib = outgoing ? ((size_t)(j * Rn + kb) * Ck + c)
                             : ((size_t)(kb * Rn + j) * Ck + c);
        b[e] = (_Float16)Rt[ib];
      }
      acc[s] = wmma16(a, b, acc[s]);
    }
  }
#pragma unroll
  for (int s = 0; s < 4; s++)
#pragma unroll
    for (int r = 0; r < 8; r++) {
      int i = it * 16 + r + 8 * half, j = (jt0 + s) * 16 + l16;
      Ot[((size_t)i * Rn + j) * Ck + c] = acc[s][r];
    }
}

// ------------------- fused outer-product-mean kernel -----------------------
// block = (i, 16 j's). Phase1: o[c,(j,d)] = sum_s left[s,i,c]*right[s,j,d]
// (one A fragment -> 8 WMMAs) -> f16 in LDS. Phase2: out[j,e] = sum_{cd} o*W
// via WMMA from LDS. pair[i,j,e] += (out + out_b[e]) / (mask_i.mask_j + 1e-3)
__global__ void k_opm(const float* __restrict__ left, const float* __restrict__ right,
                      const _Float16* __restrict__ Wp, const float* __restrict__ outb,
                      const float* __restrict__ mask, float* __restrict__ pair,
                      int Sn, int Rn, int CZn) {
  __shared__ _Float16 oLds[32][16][32];  // [c][j][d] 32KB
  __shared__ float nrm[16];
  int i = blockIdx.x, j0 = blockIdx.y * 16;
  int wave = threadIdx.y, lane = threadIdx.x;
  int half = lane >> 4, l16 = lane & 15;
  if (wave == 0 && lane < 16) {
    float s = 0.f;
    for (int ss = 0; ss < Sn; ss++) s += mask[ss * Rn + i] * mask[ss * Rn + j0 + lane];
    nrm[lane] = s + 1e-3f;
  }
  // phase 1: wave covers (mt = wave&1, 8 consecutive n-tiles)
  int mt = wave & 1, ntb = (wave >> 1) * 8;
  v8f acc[8] = {};
  for (int k0 = 0; k0 < Sn; k0 += 32) {
    v16h a;
#pragma unroll
    for (int e = 0; e < 16; e++) {
      int sa = k0 + kmapA(e, half);
      a[e] = (_Float16)left[((size_t)sa * Rn + i) * 32 + mt * 16 + l16];
    }
#pragma unroll
    for (int t = 0; t < 8; t++) {
      v16h b;
#pragma unroll
      for (int e = 0; e < 16; e++) {
        int sb = k0 + kmapB(e, half);
        b[e] = (_Float16)right[(size_t)sb * Rn * 32 + (size_t)j0 * 32 + (ntb + t) * 16 + l16];
      }
      acc[t] = wmma16(a, b, acc[t]);
    }
  }
#pragma unroll
  for (int t = 0; t < 8; t++) {
    int jd = (ntb + t) * 16 + l16;
#pragma unroll
    for (int r = 0; r < 8; r++) {
      int cc = mt * 16 + r + 8 * half;
      oLds[cc][jd >> 5][jd & 31] = (_Float16)acc[t][r];
    }
  }
  __syncthreads();
  // phase 2: K = 1024 over (c,d); Wp packed [e][1024] K-contiguous.
  for (int nt = wave; nt < (CZn >> 4); nt += 8) {
    v8f a2 = {};
    const _Float16* bp = Wp + (size_t)(nt * 16 + l16) * 1024;
    for (int k0 = 0; k0 < 1024; k0 += 32) {
      v16h a, b;
#pragma unroll
      for (int e = 0; e < 16; e++) {
        int kk = k0 + kmapA(e, half);
        a[e] = oLds[kk >> 5][l16][kk & 31];
      }
      b = *reinterpret_cast<const v16h*>(bp + k0 + (half << 4));
      a2 = wmma16(a, b, a2);
    }
    int ecol = nt * 16 + l16;
#pragma unroll
    for (int r = 0; r < 8; r++) {
      int j = j0 + r + 8 * half;
      float v = (a2[r] + outb[ecol]) / nrm[r + 8 * half];
      pair[((size_t)i * Rn + j) * CZn + ecol] += v;
    }
  }
}

// ============================ host-side helpers =============================
template <int ACT, int EPI>
static void gemm(hipStream_t st, const float* A, const _Float16* Bp, const float* bias,
                 const float* rs, float* C, float alpha, int M, int N, int Kd) {
  int Ntiles = (N + 15) >> 4;
  int blocks = (M >> 7) * ((Ntiles + 3) >> 2);  // 128x64 macro-tile per block
  k_gemm<ACT, EPI><<<dim3(blocks), dim3(32, 8), 0, st>>>(A, Bp, bias, rs, C, alpha, M, N, Kd);
}

static void layernorm(hipStream_t st, const float* x, const float* g, const float* b,
                      float* y, int rows, int C) {
  k_layernorm<<<dim3((rows + 3) / 4), dim3(32, 4), 0, st>>>(x, g, b, y, rows, C);
}

static void attention(hipStream_t st, const float* X, const float* mask, const float* nb,
                      int Bn, int N, int C, int H,
                      const _Float16* pq, const _Float16* pk, const _Float16* pv,
                      const _Float16* pg, const float* gate_b,
                      const _Float16* po, const float* out_b,
                      float* Qb, float* Kb, float* Vb, float* Gb, float* Ob, float* Lb,
                      float* target, bool addResid) {
  const int HD = H * D_;
  const float scale = 0.17677669529663689f;  // 32^-0.5
  gemm<0, 0>(st, X, pq, nullptr, nullptr, Qb, scale, Bn * N, HD, C);
  gemm<0, 0>(st, X, pk, nullptr, nullptr, Kb, 1.f, Bn * N, HD, C);
  gemm<0, 0>(st, X, pv, nullptr, nullptr, Vb, 1.f, Bn * N, HD, C);
  gemm<2, 0>(st, X, pg, gate_b, nullptr, Gb, 1.f, Bn * N, HD, C);
  const int Bc = 16, nt16 = N >> 4;
  for (int b0 = 0; b0 < Bn; b0 += Bc) {
    int tq = Bc * H * nt16 * (nt16 >> 2);
    k_qkt<<<dim3((tq + 7) / 8), dim3(32, 8), 0, st>>>(Qb, Kb, Lb, b0, Bc, N, H, HD);
    int rows = Bc * H * N;
    k_softmax<<<dim3((rows + 7) / 8), dim3(32, 8), 0, st>>>(Lb, mask, nb, b0, Bc, N, H);
    int ta = Bc * H * nt16;
    k_av<<<dim3((ta + 7) / 8), dim3(32, 8), 0, st>>>(Lb, Vb, Ob, b0, Bc, N, H, HD);
  }
  size_t n = (size_t)Bn * N * HD;
  k_mul<<<dim3((int)((n + 255) / 256)), dim3(256), 0, st>>>(Ob, Gb, n);
  if (addResid) gemm<0, 1>(st, Ob, po, out_b, nullptr, target, 1.f, Bn * N, C, HD);
  else          gemm<0, 0>(st, Ob, po, out_b, nullptr, target, 1.f, Bn * N, C, HD);
}

// -------------------------- input index mapping ----------------------------
enum {
  IN_MSA = 0, IN_PAIR = 1, IN_MSA_MASK = 2, IN_PAIR_MASK = 3,
  COL_GATE_B = 4, COL_GATE_W, COL_K_W, COL_LN_B, COL_LN_G, COL_OUT_B, COL_OUT_W, COL_Q_W, COL_V_W,
  MT_B1 = 13, MT_B2, MT_LN_B, MT_LN_G, MT_W1, MT_W2,
  OPM_LEFT_B = 19, OPM_LEFT_W, OPM_LN_B, OPM_LN_G, OPM_OUT_B, OPM_OUT_W, OPM_RIGHT_B, OPM_RIGHT_W,
  PT_B1 = 27, PT_B2, PT_LN_B, PT_LN_G, PT_W1, PT_W2,
  ROW_FEAT2D = 33, ROW_GATE_B, ROW_GATE_W, ROW_K_W, ROW_LN_B, ROW_LN_G, ROW_OUT_B, ROW_OUT_W,
  ROW_PAIR_LN_B, ROW_PAIR_LN_G, ROW_Q_W, ROW_V_W,
  TAE_BASE = 45, TAS_BASE = 55, TMI_BASE = 65, TMO_BASE = 81
};
// relative offsets within a tri-attention block (sorted keys)
enum { TA_FEAT = 0, TA_GATE_B, TA_GATE_W, TA_K_W, TA_LN_B, TA_LN_G, TA_OUT_B, TA_OUT_W, TA_Q_W, TA_V_W };
// relative offsets within a tri-mult block (sorted keys)
enum { TM_CLN_B = 0, TM_CLN_G, TM_GATE_B, TM_GATE_W, TM_LEFT_B, TM_LGATE_B, TM_LGATE_W, TM_LEFT_W,
       TM_LN_B, TM_LN_G, TM_OUT_B, TM_OUT_W, TM_RIGHT_B, TM_RGATE_B, TM_RGATE_W, TM_RIGHT_W };

extern "C" void kernel_launch(void* const* d_in, const int* in_sizes, int n_in,
                              void* d_out, int out_size, void* d_ws, size_t ws_size,
                              hipStream_t stream) {
  (void)in_sizes; (void)out_size; (void)ws_size;
  if (n_in < 97) return;
  auto P = [&](int i) -> const float* { return (const float*)d_in[i]; };

  float* msa  = (float*)d_out;          // residual accumulator [S,R,CM]
  float* pair = (float*)d_out + MS_;    // residual accumulator [R,R,CZ]
  float* w = (float*)d_ws;
  float *X = w + OFF_X, *Z = w + OFF_Z, *Qb = w + OFF_Q, *Kb = w + OFF_K,
        *Vb = w + OFF_V, *Gb = w + OFF_G, *Ob = w + OFF_O, *Lb = w + OFF_L,
        *Hb = w + OFF_H, *NB = w + OFF_NB, *MT = w + OFF_MT;
  _Float16* pkBase = (_Float16*)(w + OFF_PK);

  hipMemcpyAsync(msa,  P(IN_MSA),  MS_ * sizeof(float), hipMemcpyDeviceToDevice, stream);
  hipMemcpyAsync(pair, P(IN_PAIR), PS_ * sizeof(float), hipMemcpyDeviceToDevice, stream);

  size_t pkOfs = 0;
  auto pack = [&](const float* Bsrc, int Kd, int N) -> const _Float16* {
    int Npad = (N + 63) & ~63;  // 64-col padding for the cooperative copy
    _Float16* dst = pkBase + pkOfs;
    pkOfs += (size_t)Npad * Kd;
    int total = Npad * Kd;
    k_pack<<<dim3((total + 255) / 256), dim3(256), 0, stream>>>(Bsrc, dst, Kd, N, Npad);
    return dst;
  };

  // ---------------- pack static (non-tri) weights --------------------------
  const _Float16 *pRowF = pack(P(ROW_FEAT2D), CZ_, HM_);
  const _Float16 *pRowQ = pack(P(ROW_Q_W), CM_, HM_ * D_);
  const _Float16 *pRowK = pack(P(ROW_K_W), CM_, HM_ * D_);
  const _Float16 *pRowV = pack(P(ROW_V_W), CM_, HM_ * D_);
  const _Float16 *pRowG = pack(P(ROW_GATE_W), CM_, HM_ * D_);
  const _Float16 *pRowO = pack(P(ROW_OUT_W), HM_ * D_, CM_);
  const _Float16 *pColQ = pack(P(COL_Q_W), CM_, HM_ * D_);
  const _Float16 *pColK = pack(P(COL_K_W), CM_, HM_ * D_);
  const _Float16 *pColV = pack(P(COL_V_W), CM_, HM_ * D_);
  const _Float16 *pColG = pack(P(COL_GATE_W), CM_, HM_ * D_);
  const _Float16 *pColO = pack(P(COL_OUT_W), HM_ * D_, CM_);
  const _Float16 *pMTW1 = pack(P(MT_W1), CM_, 4 * CM_);
  const _Float16 *pMTW2 = pack(P(MT_W2), 4 * CM_, CM_);
  const _Float16 *pOpL  = pack(P(OPM_LEFT_W), CM_, 32);
  const _Float16 *pOpR  = pack(P(OPM_RIGHT_W), CM_, 32);
  const _Float16 *pOpO  = pack(P(OPM_OUT_W), 1024, CZ_);   // [COP*COP, CZ]
  const _Float16 *pPTW1 = pack(P(PT_W1), CZ_, 4 * CZ_);
  const _Float16 *pPTW2 = pack(P(PT_W2), 4 * CZ_, CZ_);

  // ---------------- 1. MSA row attention (pair-biased, gated) --------------
  layernorm(stream, msa, P(ROW_LN_G), P(ROW_LN_B), X, (int)SR_, CM_);
  layernorm(stream, pair, P(ROW_PAIR_LN_G), P(ROW_PAIR_LN_B), Z, (int)RR_, CZ_);
  gemm<0, 0>(stream, Z, pRowF, nullptr, nullptr, NB, 1.f, (int)RR_, HM_, CZ_);
  attention(stream, X, P(IN_MSA_MASK), NB, S_, R_, CM_, HM_,
            pRowQ, pRowK, pRowV, pRowG, P(ROW_GATE_B), pRowO, P(ROW_OUT_B),
            Qb, Kb, Vb, Gb, Ob, Lb, msa, true);

  // ---------------- 2. MSA column attention --------------------------------
  k_transpose01<<<dim3((int)((MS_ + 255) / 256)), dim3(256), 0, stream>>>(msa, Z, S_, R_, CM_);
  k_transpose01<<<dim3((int)((SR_ + 255) / 256)), dim3(256), 0, stream>>>(P(IN_MSA_MASK), MT, S_, R_, 1);
  layernorm(stream, Z, P(COL_LN_G), P(COL_LN_B), X, (int)SR_, CM_);
  attention(stream, X, MT, nullptr, R_, S_, CM_, HM_,
            pColQ, pColK, pColV, pColG, P(COL_GATE_B), pColO, P(COL_OUT_B),
            Qb, Kb, Vb, Gb, Ob, Lb, Z, false);
  k_transpose_add<<<dim3((int)((MS_ + 255) / 256)), dim3(256), 0, stream>>>(Z, msa, S_, R_, CM_);

  // ---------------- 3. MSA transition --------------------------------------
  layernorm(stream, msa, P(MT_LN_G), P(MT_LN_B), X, (int)SR_, CM_);
  gemm<1, 0>(stream, X, pMTW1, P(MT_B1), nullptr, Hb, 1.f, (int)SR_, 4 * CM_, CM_);
  gemm<0, 1>(stream, Hb, pMTW2, P(MT_B2), nullptr, msa, 1.f, (int)SR_, CM_, 4 * CM_);

  // ---------------- 4. Outer product mean ----------------------------------
  layernorm(stream, msa, P(OPM_LN_G), P(OPM_LN_B), X, (int)SR_, CM_);
  gemm<0, 0>(stream, X, pOpL, P(OPM_LEFT_B), P(IN_MSA_MASK), Qb, 1.f, (int)SR_, 32, CM_);
  gemm<0, 0>(stream, X, pOpR, P(OPM_RIGHT_B), P(IN_MSA_MASK), Kb, 1.f, (int)SR_, 32, CM_);
  k_opm<<<dim3(R_, R_ / 16), dim3(32, 8), 0, stream>>>(
      Qb, Kb, pOpO, P(OPM_OUT_B), P(IN_MSA_MASK), pair, S_, R_, CZ_);

  // ---------------- 5/6. Triangle multiplications --------------------------
  auto tri_mul_f = [&](int base, bool outgoing) {
    const _Float16* pl  = pack(P(base + TM_LEFT_W), CZ_, CT_);
    const _Float16* plg = pack(P(base + TM_LGATE_W), CZ_, CT_);
    const _Float16* pr  = pack(P(base + TM_RIGHT_W), CZ_, CT_);
    const _Float16* prg = pack(P(base + TM_RGATE_W), CZ_, CT_);
    const _Float16* po2 = pack(P(base + TM_OUT_W), CT_, CZ_);
    const _Float16* pgt = pack(P(base + TM_GATE_W), CZ_, CZ_);
    layernorm(stream, pair, P(base + TM_LN_G), P(base + TM_LN_B), X, (int)RR_, CZ_);
    gemm<0, 0>(stream, X, pl,  P(base + TM_LEFT_B),  P(IN_PAIR_MASK), Qb, 1.f, (int)RR_, CT_, CZ_);
    gemm<2, 2>(stream, X, plg, P(base + TM_LGATE_B), nullptr,         Qb, 1.f, (int)RR_, CT_, CZ_);
    gemm<0, 0>(stream, X, pr,  P(base + TM_RIGHT_B), P(IN_PAIR_MASK), Kb, 1.f, (int)RR_, CT_, CZ_);
    gemm<2, 2>(stream, X, prg, P(base + TM_RGATE_B), nullptr,         Kb, 1.f, (int)RR_, CT_, CZ_);
    k_trimul_o<<<dim3(R_ / 16, R_ / 64, CT_), dim3(32), 0, stream>>>(Qb, Kb, Vb, R_, CT_, outgoing ? 1 : 0);
    layernorm(stream, Vb, P(base + TM_CLN_G), P(base + TM_CLN_B), Gb, (int)RR_, CT_);
    gemm<0, 0>(stream, Gb, po2, P(base + TM_OUT_B), nullptr, Qb, 1.f, (int)RR_, CZ_, CT_);
    gemm<2, 0>(stream, X, pgt, P(base + TM_GATE_B), nullptr, Kb, 1.f, (int)RR_, CZ_, CZ_);
    k_fma_add<<<dim3((int)((PS_ + 255) / 256)), dim3(256), 0, stream>>>(pair, Qb, Kb, PS_);
  };
  tri_mul_f(TMO_BASE, true);    // outgoing
  tri_mul_f(TMI_BASE, false);   // incoming

  // ---------------- 7/8. Triangle attentions -------------------------------
  auto tri_attn_f = [&](int base, bool starting) {
    const _Float16* pF = pack(P(base + TA_FEAT), CZ_, HZ_);
    const _Float16* pQ = pack(P(base + TA_Q_W), CZ_, HZ_ * D_);
    const _Float16* pK = pack(P(base + TA_K_W), CZ_, HZ_ * D_);
    const _Float16* pV = pack(P(base + TA_V_W), CZ_, HZ_ * D_);
    const _Float16* pG = pack(P(base + TA_GATE_W), CZ_, HZ_ * D_);
    const _Float16* pO = pack(P(base + TA_OUT_W), HZ_ * D_, CZ_);
    const float* xin;
    const float* maskp;
    if (starting) {
      xin = pair; maskp = P(IN_PAIR_MASK);
    } else {
      k_transpose01<<<dim3((int)((PS_ + 255) / 256)), dim3(256), 0, stream>>>(pair, Z, R_, R_, CZ_);
      k_transpose01<<<dim3((int)((RR_ + 255) / 256)), dim3(256), 0, stream>>>(P(IN_PAIR_MASK), MT, R_, R_, 1);
      xin = Z; maskp = MT;
    }
    layernorm(stream, xin, P(base + TA_LN_G), P(base + TA_LN_B), X, (int)RR_, CZ_);
    gemm<0, 0>(stream, X, pF, nullptr, nullptr, NB, 1.f, (int)RR_, HZ_, CZ_);
    if (starting) {
      attention(stream, X, maskp, NB, R_, R_, CZ_, HZ_, pQ, pK, pV, pG,
                P(base + TA_GATE_B), pO, P(base + TA_OUT_B),
                Qb, Kb, Vb, Gb, Ob, Lb, pair, true);
    } else {
      attention(stream, X, maskp, NB, R_, R_, CZ_, HZ_, pQ, pK, pV, pG,
                P(base + TA_GATE_B), pO, P(base + TA_OUT_B),
                Qb, Kb, Vb, Gb, Ob, Lb, Z, false);
      k_transpose_add<<<dim3((int)((PS_ + 255) / 256)), dim3(256), 0, stream>>>(Z, pair, R_, R_, CZ_);
    }
  };
  tri_attn_f(TAS_BASE, true);
  tri_attn_f(TAE_BASE, false);

  // ---------------- 9. Pair transition -------------------------------------
  layernorm(stream, pair, P(PT_LN_G), P(PT_LN_B), X, (int)RR_, CZ_);
  gemm<1, 0>(stream, X, pPTW1, P(PT_B1), nullptr, Hb, 1.f, (int)RR_, 4 * CZ_, CZ_);
  gemm<0, 1>(stream, Hb, pPTW2, P(PT_B2), nullptr, pair, 1.f, (int)RR_, CZ_, 4 * CZ_);
}
